// AVnet_Runtime_47210280517791
// MI455X (gfx1250) — matrix operand order repaired and checked by reading, not compiled, not run on hardware
//
#include <hip/hip_runtime.h>
#include <math.h>

// ---------------------------------------------------------------------------
// AVnet runtime forward for MI455X (gfx1250, wave32, WMMA f16 16x16x32).
// fp32 storage; f16 conversion at LDS-staging time; f32 accumulation.
// GEMM: double-buffered LDS software pipeline (global loads overlap WMMA).
// ---------------------------------------------------------------------------

typedef __attribute__((ext_vector_type(16))) _Float16 v16h;
typedef __attribute__((ext_vector_type(8)))  _Float16 v8h;
typedef __attribute__((ext_vector_type(4)))  _Float16 v4h;
typedef __attribute__((ext_vector_type(8)))  float    v8f;
typedef __attribute__((ext_vector_type(4)))  float    v4f;

#define B_     16
#define D_     768
#define H_     12
#define DH_    64
#define NAUD_  512
#define NIMG_  196
#define NP_    708      // NAUD_ + NIMG_
#define KKEEP_ 495      // int(0.7 * 708)
#define NTA_   496      // kept audio tokens + cls
#define NTI_   197      // kept image tokens + cls
#define FEATN_ 691      // 495 + 196
#define NCLS_  309
#define EPS_   1e-6f
#define SCALE_ 0.125f   // DH^-0.5

__device__ inline float gelu_f(float x) {
  return 0.5f * x * (1.0f + erff(x * 0.70710678118654752f));
}

// A-fragment (16x32 f16, M x K): lane holds row M = lane%16.
// lanes 0-15: K = 0..15 ; lanes 16-31: K = 16..31.
__device__ inline v16h mk_a_frag(const _Float16* rowp, int hs) {
  return *(const v16h*)(rowp + hs * 16);
}
// B-fragment (32x16 f16, K x N): lane holds col N = lane%16.
// lanes 0-15: K = 0..15 ; lanes 16-31: K = 16..31. LDS stored [n][k].
__device__ inline v16h mk_b_frag(const _Float16* colp, int hs) {
  return *(const v16h*)(colp + hs * 16);
}
__device__ inline v8f wmma32(v16h a, v16h b, v8f c) {
  return __builtin_amdgcn_wmma_f32_16x16x32_f16(false, a, false, b, (short)0, c,
                                                false, false);
}
__device__ inline v4h cvt4(v4f v, float f) {
  v4h h;
  h.x = (_Float16)(v.x * f); h.y = (_Float16)(v.y * f);
  h.z = (_Float16)(v.z * f); h.w = (_Float16)(v.w * f);
  return h;
}

// ---------------------------------------------------------------------------
// Generic WMMA GEMM: C[M,N] = act(A[M,K] @ W[K,N] + bias) (+ res)
// 256 threads = 8 waves (2x4), 128x128 tile, K step 32 (K % 32 == 0).
// Double-buffered LDS pipeline; branchless clamped staging.
// act: 0 = none, 1 = exact gelu. res may alias C (read-before-write per elem).
// ---------------------------------------------------------------------------
__global__ __launch_bounds__(256) void k_gemm(
    const float* __restrict__ A, const float* __restrict__ W,
    const float* __restrict__ bias, const float* __restrict__ res,
    float* __restrict__ C, int M, int N, int K, int act)
{
  __shared__ __align__(32) _Float16 sA[2][128][48];  // [buf][m][k], pad 16
  __shared__ __align__(32) _Float16 sB[2][128][48];  // [buf][n][k]
  const int tid = threadIdx.x, lane = tid & 31;
  const int wv = tid >> 5, wr = wv >> 2, wc = wv & 3;
  const int m0 = blockIdx.y * 128, n0 = blockIdx.x * 128;
  const int lm = lane & 15, hs = (lane < 16) ? 0 : 1;
  const bool bFast = (n0 + 128 <= N) && ((N & 3) == 0);  // block-uniform
  const int nT = K >> 5;

  // ---- loop-invariant staging coordinates ----
  int aO[4]; const float* aP[4]; float aF[4];
#pragma unroll
  for (int i = 0; i < 4; ++i) {
    int e4 = tid + 256 * i;            // 0..1023 vec4 groups of the A tile
    int m = e4 >> 3, k = (e4 & 7) * 4;
    int gm = m0 + m;
    int gmc = (gm < M) ? gm : (M - 1);
    aO[i] = m * 48 + k;
    aP[i] = A + (size_t)gmc * K + k;
    aF[i] = (gm < M) ? 1.f : 0.f;
  }
  int bO[4]; const float* bP[4];
  if (bFast) {
#pragma unroll
    for (int i = 0; i < 4; ++i) {
      int e4 = tid + 256 * i;
      int k = e4 >> 5, n = (e4 & 31) * 4;
      bO[i] = n * 48 + k;
      bP[i] = W + (size_t)k * N + n0 + n;
    }
  }

  const v8f vzero = {0.f,0.f,0.f,0.f,0.f,0.f,0.f,0.f};
  v8f acc[4][2];
  for (int i = 0; i < 4; ++i)
    for (int j = 0; j < 2; ++j) acc[i][j] = vzero;

  v4f aR[4];
  float bReg[16];

  // ---- prologue: load + store tile 0 into buffer 0 ----
#pragma unroll
  for (int i = 0; i < 4; ++i) aR[i] = *(const v4f*)(aP[i]);
  if (bFast) {
#pragma unroll
    for (int i = 0; i < 4; ++i) *(v4f*)&bReg[4 * i] = *(const v4f*)(bP[i]);
  } else {
#pragma unroll
    for (int i = 0; i < 16; ++i) {
      int e = tid + 256 * i;
      int k = e >> 7, n = e & 127;
      int gn = n0 + n;
      int gnc = (gn < N) ? gn : (N - 1);
      bReg[i] = W[(size_t)k * N + gnc];
    }
  }
#pragma unroll
  for (int i = 0; i < 4; ++i) *(v4h*)&sA[0][0][aO[i]] = cvt4(aR[i], aF[i]);
  if (bFast) {
#pragma unroll
    for (int i = 0; i < 4; ++i) {
      v4h hv = cvt4(*(v4f*)&bReg[4 * i], 1.f);
      sB[0][0][bO[i] + 0 * 48] = hv.x;
      sB[0][0][bO[i] + 1 * 48] = hv.y;
      sB[0][0][bO[i] + 2 * 48] = hv.z;
      sB[0][0][bO[i] + 3 * 48] = hv.w;
    }
  } else {
#pragma unroll
    for (int i = 0; i < 16; ++i) {
      int e = tid + 256 * i;
      int k = e >> 7, n = e & 127;
      int gn = n0 + n;
      sB[0][n][k] = (_Float16)((gn < N) ? bReg[i] : 0.f);
    }
  }
  __syncthreads();

  // ---- pipelined main loop: one barrier per K-step ----
  for (int t = 0; t < nT; ++t) {
    const int cur = t & 1;
    const bool hasNext = (t + 1) < nT;
    const int kk = (t + 1) << 5;
    if (hasNext) {                         // issue next-tile loads (in flight
#pragma unroll                             //  under this tile's WMMAs)
      for (int i = 0; i < 4; ++i) aR[i] = *(const v4f*)(aP[i] + kk);
      if (bFast) {
#pragma unroll
        for (int i = 0; i < 4; ++i)
          *(v4f*)&bReg[4 * i] = *(const v4f*)(bP[i] + (size_t)kk * N);
      } else {
#pragma unroll
        for (int i = 0; i < 16; ++i) {
          int e = tid + 256 * i;
          int k = e >> 7, n = e & 127;
          int gn = n0 + n;
          int gnc = (gn < N) ? gn : (N - 1);
          bReg[i] = W[(size_t)(kk + k) * N + gnc];
        }
      }
    }
    // compute tile t from buffer `cur`
    const _Float16* sAc = &sA[cur][0][0];
    const _Float16* sBc = &sB[cur][0][0];
#pragma unroll
    for (int mi = 0; mi < 4; ++mi) {
      v16h a = mk_a_frag(sAc + (wr * 64 + mi * 16 + lm) * 48, hs);
#pragma unroll
      for (int ni = 0; ni < 2; ++ni) {
        v16h b = mk_b_frag(sBc + (wc * 32 + ni * 16 + lm) * 48, hs);
        acc[mi][ni] = wmma32(a, b, acc[mi][ni]);
      }
    }
    if (hasNext) {                         // convert + store into other buffer
      const int nxt = cur ^ 1;
      _Float16* sAn = &sA[nxt][0][0];
      _Float16* sBn = &sB[nxt][0][0];
#pragma unroll
      for (int i = 0; i < 4; ++i) *(v4h*)(sAn + aO[i]) = cvt4(aR[i], aF[i]);
      if (bFast) {
#pragma unroll
        for (int i = 0; i < 4; ++i) {
          v4h hv = cvt4(*(v4f*)&bReg[4 * i], 1.f);
          sBn[bO[i] + 0 * 48] = hv.x;
          sBn[bO[i] + 1 * 48] = hv.y;
          sBn[bO[i] + 2 * 48] = hv.z;
          sBn[bO[i] + 3 * 48] = hv.w;
        }
      } else {
#pragma unroll
        for (int i = 0; i < 16; ++i) {
          int e = tid + 256 * i;
          int k = e >> 7, n = e & 127;
          int gn = n0 + n;
          sBn[n * 48 + k] = (_Float16)((gn < N) ? bReg[i] : 0.f);
        }
      }
    }
    __syncthreads();
  }

  // ---- epilogue: C/D layout (VGPR r: rows r / r+8 by lane half) ----
  const int rbase = m0 + wr * 64 + ((lane < 16) ? 0 : 8);
  const int cbase = n0 + wc * 32 + lm;
  if ((m0 + 128 <= M) && (n0 + 128 <= N)) {   // interior: unguarded stores
    for (int mi = 0; mi < 4; ++mi)
      for (int ni = 0; ni < 2; ++ni) {
#pragma unroll
        for (int r = 0; r < 8; ++r) {
          int row = rbase + mi * 16 + r;
          int col = cbase + ni * 16;
          float v = acc[mi][ni][r];
          if (bias) v += bias[col];
          if (act == 1) v = gelu_f(v);
          if (res) v += res[(size_t)row * N + col];
          C[(size_t)row * N + col] = v;
        }
      }
  } else {
    for (int mi = 0; mi < 4; ++mi)
      for (int ni = 0; ni < 2; ++ni) {
#pragma unroll
        for (int r = 0; r < 8; ++r) {
          int row = rbase + mi * 16 + r;
          int col = cbase + ni * 16;
          if (row < M && col < N) {
            float v = acc[mi][ni][r];
            if (bias) v += bias[col];
            if (act == 1) v = gelu_f(v);
            if (res) v += res[(size_t)row * N + col];
            C[(size_t)row * N + col] = v;
          }
        }
      }
  }
}

// ---------------------------------------------------------------------------
// LayerNorm, one block per row.
// ---------------------------------------------------------------------------
__global__ __launch_bounds__(256) void k_layernorm(
    const float* __restrict__ x, const float* __restrict__ g,
    const float* __restrict__ bb, float* __restrict__ y, int C)
{
  __shared__ float red[256];
  const size_t row = blockIdx.x;
  const float* xr = x + row * (size_t)C;
  const int tid = threadIdx.x;
  float s = 0.f;
  for (int c = tid; c < C; c += 256) s += xr[c];
  red[tid] = s; __syncthreads();
  for (int o = 128; o > 0; o >>= 1) { if (tid < o) red[tid] += red[tid + o]; __syncthreads(); }
  float mean = red[0] / (float)C;
  __syncthreads();
  float s2 = 0.f;
  for (int c = tid; c < C; c += 256) { float d = xr[c] - mean; s2 += d * d; }
  red[tid] = s2; __syncthreads();
  for (int o = 128; o > 0; o >>= 1) { if (tid < o) red[tid] += red[tid + o]; __syncthreads(); }
  float rs = rsqrtf(red[0] / (float)C + EPS_);
  float* yr = y + row * (size_t)C;
  for (int c = tid; c < C; c += 256) yr[c] = (xr[c] - mean) * rs * g[c] + bb[c];
}

// ---------------------------------------------------------------------------
// Fused attention with policy-softmax.  grid = (ceil(Nt/32), H, B), 8 waves.
// qkv row layout per token: [q(768) | k(768) | v(768)], head h at h*64.
// Dynamic LDS: sQ 32x64 | sK Npad x 64 ([t][d]) | sVt 64 x Npad ([d][t])
//            | sS 32 x Npad f32 | sP 32 x Npad f16.  Npad % 64 == 0.
// ---------------------------------------------------------------------------
__global__ __launch_bounds__(256) void k_attn(
    const float* __restrict__ qkv, const float* __restrict__ pol,
    float* __restrict__ o, int Nt, int Npad)
{
  extern __shared__ char smem[];
  _Float16* sQ  = (_Float16*)smem;                     // 32*64
  _Float16* sK  = sQ + 32 * 64;                        // Npad*64
  _Float16* sVt = sK + (size_t)Npad * 64;              // 64*Npad
  float*    sS  = (float*)(sVt + (size_t)64 * Npad);   // 32*Npad
  _Float16* sP  = (_Float16*)(sS + (size_t)32 * Npad); // 32*Npad

  const int b = blockIdx.z, h = blockIdx.y;
  const int row0 = blockIdx.x * 32;
  const int tid = threadIdx.x, lane = tid & 31, wv = tid >> 5;
  const int lm = lane & 15, hs = (lane < 16) ? 0 : 1;
  const float* base = qkv + (size_t)b * Nt * (3 * D_);
  const v8f vzero = {0.f,0.f,0.f,0.f,0.f,0.f,0.f,0.f};

  // Q (pre-scaled): 32x64, two batched vec4 loads then stores
  {
    v4f v[2]; int off[2]; float ff[2];
#pragma unroll
    for (int u = 0; u < 2; ++u) {
      int e4 = tid + 256 * u;            // 0..511
      int m = e4 >> 4, d = (e4 & 15) * 4;
      int gr = row0 + m;
      int grc = (gr < Nt) ? gr : (Nt - 1);
      v[u] = *(const v4f*)(base + (size_t)grc * (3 * D_) + h * DH_ + d);
      off[u] = m * 64 + d;
      ff[u] = (gr < Nt) ? SCALE_ : 0.f;
    }
#pragma unroll
    for (int u = 0; u < 2; ++u) *(v4h*)&sQ[off[u]] = cvt4(v[u], ff[u]);
  }
  // K as [t][d]: batches of 4 vec4 loads (Npad*16 % 1024 == 0)
  for (int eb = tid; eb < Npad * 16; eb += 1024) {
    v4f v[4]; int off[4]; float ff[4];
#pragma unroll
    for (int u = 0; u < 4; ++u) {
      int e4 = eb + 256 * u;
      int t = e4 >> 4, d = (e4 & 15) * 4;
      int tc = (t < Nt) ? t : (Nt - 1);
      v[u] = *(const v4f*)(base + (size_t)tc * (3 * D_) + D_ + h * DH_ + d);
      off[u] = t * 64 + d;
      ff[u] = (t < Nt) ? 1.f : 0.f;
    }
#pragma unroll
    for (int u = 0; u < 4; ++u) *(v4h*)&sK[off[u]] = cvt4(v[u], ff[u]);
  }
  // V transposed [d][t]: batched loads, scattered b16 stores
  for (int eb = tid; eb < Npad * 16; eb += 1024) {
    v4f v[4]; int tt[4]; int dd[4]; float ff[4];
#pragma unroll
    for (int u = 0; u < 4; ++u) {
      int e4 = eb + 256 * u;
      int t = e4 >> 4, d = (e4 & 15) * 4;
      int tc = (t < Nt) ? t : (Nt - 1);
      v[u] = *(const v4f*)(base + (size_t)tc * (3 * D_) + 2 * D_ + h * DH_ + d);
      tt[u] = t; dd[u] = d;
      ff[u] = (t < Nt) ? 1.f : 0.f;
    }
#pragma unroll
    for (int u = 0; u < 4; ++u) {
      sVt[(size_t)(dd[u] + 0) * Npad + tt[u]] = (_Float16)(v[u].x * ff[u]);
      sVt[(size_t)(dd[u] + 1) * Npad + tt[u]] = (_Float16)(v[u].y * ff[u]);
      sVt[(size_t)(dd[u] + 2) * Npad + tt[u]] = (_Float16)(v[u].z * ff[u]);
      sVt[(size_t)(dd[u] + 3) * Npad + tt[u]] = (_Float16)(v[u].w * ff[u]);
    }
  }
  __syncthreads();

  // ---- phase 1: S = (Q*scale) @ K^T, each wave owns a 64-col chunk ----
  {
    const int c0 = wv * 64;
    if (c0 < Npad) {                                   // wave-uniform
      v8f acc[2][4];
      for (int i = 0; i < 2; ++i)
        for (int j = 0; j < 4; ++j) acc[i][j] = vzero;
      for (int kk = 0; kk < 64; kk += 32) {
#pragma unroll
        for (int mi = 0; mi < 2; ++mi) {
          v16h a = mk_a_frag(&sQ[(mi * 16 + lm) * 64 + kk], hs);
#pragma unroll
          for (int nj = 0; nj < 4; ++nj) {
            v16h bf = mk_b_frag(&sK[(size_t)(c0 + nj * 16 + lm) * 64 + kk], hs);
            acc[mi][nj] = wmma32(a, bf, acc[mi][nj]);
          }
        }
      }
      for (int mi = 0; mi < 2; ++mi)
        for (int nj = 0; nj < 4; ++nj) {
#pragma unroll
          for (int r = 0; r < 8; ++r) {
            int rr = mi * 16 + r + ((lane < 16) ? 0 : 8);
            sS[(size_t)rr * Npad + c0 + nj * 16 + lm] = acc[mi][nj][r];
          }
        }
    }
  }
  __syncthreads();

  // ---- phase 2: policy softmax, 8 threads per row ----
  {
    const int r = tid >> 3, g = tid & 7;
    const int gr = row0 + r;
    const float* prow = pol + (size_t)b * Nt;
    float mx = -3.4e38f;
    if (gr < Nt)
      for (int c = g; c < Nt; c += 8) mx = fmaxf(mx, sS[(size_t)r * Npad + c]);
    mx = fmaxf(mx, __shfl_xor(mx, 1, 8));
    mx = fmaxf(mx, __shfl_xor(mx, 2, 8));
    mx = fmaxf(mx, __shfl_xor(mx, 4, 8));
    float sum = 0.f;
    if (gr < Nt)
      for (int c = g; c < Nt; c += 8) {
        float P = prow[c];
        float pv = P + (1.f - P) * ((c == gr) ? 1.f : 0.f);
        sum += expf(sS[(size_t)r * Npad + c] - mx) * pv;
      }
    sum += __shfl_xor(sum, 1, 8);
    sum += __shfl_xor(sum, 2, 8);
    sum += __shfl_xor(sum, 4, 8);
    const float inv = 1.f / (sum + EPS_);
    const float addc = EPS_ / (float)Nt;
    for (int c = g; c < Npad; c += 8) {
      float p = 0.f;
      if (gr < Nt && c < Nt) {
        float P = prow[c];
        float pv = P + (1.f - P) * ((c == gr) ? 1.f : 0.f);
        p = (expf(sS[(size_t)r * Npad + c] - mx) * pv + addc) * inv;
      }
      sP[(size_t)r * Npad + c] = (_Float16)p;
    }
  }
  __syncthreads();

  // ---- phase 3: O = P @ V, one 16x16 fragment per wave ----
  {
    const int mi = wv >> 2, nj = wv & 3;
    v8f acc = vzero;
    for (int kk = 0; kk < Npad; kk += 32) {
      v16h a  = mk_a_frag(&sP[(size_t)(mi * 16 + lm) * Npad + kk], hs);
      v16h bf = mk_b_frag(&sVt[(size_t)(nj * 16 + lm) * Npad + kk], hs);
      acc = wmma32(a, bf, acc);
    }
    const int rb = row0 + mi * 16 + ((lane < 16) ? 0 : 8);
    const int dc = h * DH_ + nj * 16 + lm;
    if (row0 + 32 <= Nt) {               // block-uniform fast path
#pragma unroll
      for (int r = 0; r < 8; ++r)
        o[((size_t)b * Nt + rb + r) * D_ + dc] = acc[r];
    } else {
#pragma unroll
      for (int r = 0; r < 8; ++r)
        if (rb + r < Nt) o[((size_t)b * Nt + rb + r) * D_ + dc] = acc[r];
    }
  }
}

// ---------------------------------------------------------------------------
// Elementwise / data movement helpers
// ---------------------------------------------------------------------------
__global__ void k_build_spatial(const float* __restrict__ audio,
                                const float* __restrict__ image,
                                float* __restrict__ sp)
{
  int n = blockIdx.x, b = blockIdx.y;
  const float* src = (n < NAUD_)
      ? audio + ((size_t)b * (NAUD_ + 1) + 1 + n) * D_
      : image + ((size_t)b * (NIMG_ + 1) + 1 + (n - NAUD_)) * D_;
  float* dst = sp + ((size_t)b * NP_ + n) * D_;
  for (int c = threadIdx.x; c < D_ / 4; c += 256)
    ((v4f*)dst)[c] = ((const v4f*)src)[c];
}

__global__ __launch_bounds__(256) void k_pool(const float* __restrict__ h,
                                              float* __restrict__ gpool)
{
  __shared__ float red[256];
  int c = blockIdx.x, b = blockIdx.y;
  float s = 0.f;
  for (int n = threadIdx.x; n < NP_; n += 256)
    s += h[((size_t)b * NP_ + n) * D_ + (D_ / 2) + c];
  red[threadIdx.x] = s; __syncthreads();
  for (int o = 128; o > 0; o >>= 1) {
    if (threadIdx.x < o) red[threadIdx.x] += red[threadIdx.x + o];
    __syncthreads();
  }
  if (threadIdx.x == 0) gpool[(size_t)b * (D_ / 2) + c] = red[0] / (float)NP_;
}

__global__ void k_build_h2(const float* __restrict__ h,
                           const float* __restrict__ gpool,
                           float* __restrict__ h2)
{
  int n = blockIdx.x, b = blockIdx.y;
  const float* src = h + ((size_t)b * NP_ + n) * D_;
  float* dst = h2 + ((size_t)b * NP_ + n) * D_;
  for (int c = threadIdx.x; c < D_; c += 256)
    dst[c] = (c < D_ / 2) ? src[c] : gpool[(size_t)b * (D_ / 2) + (c - D_ / 2)];
}

__global__ void k_score(const float* __restrict__ y2, float* __restrict__ score)
{
  int i = blockIdx.x * 256 + threadIdx.x;
  if (i < B_ * NP_) {
    float a0 = y2[2 * i], a1 = y2[2 * i + 1];
    float m = fmaxf(a0, a1);
    score[i] = a0 - (m + logf(expf(a0 - m) + expf(a1 - m)));  // log_softmax[...,0]
  }
}

// Per-batch: stable top-K + stable audio/image partition -> policies + gathers.
__global__ __launch_bounds__(256) void k_topk(const float* __restrict__ score,
                                              float* __restrict__ pol_a,
                                              float* __restrict__ pol_i,
                                              int* __restrict__ ga,
                                              int* __restrict__ gi)
{
  __shared__ float ss[1024];
  __shared__ int   si[1024];
  __shared__ int lista[KKEEP_];
  __shared__ int listi[NIMG_];
  __shared__ int s_na;
  const int b = blockIdx.x, tid = threadIdx.x;
  for (int i = tid; i < 1024; i += 256) {
    ss[i] = (i < NP_) ? score[(size_t)b * NP_ + i] : -3.4e38f;
    si[i] = i;
  }
  __syncthreads();
  // Bitonic sort: order = descending score, ties by ascending index
  for (int k = 2; k <= 1024; k <<= 1) {
    for (int j = k >> 1; j > 0; j >>= 1) {
      for (int i = tid; i < 1024; i += 256) {
        int ixj = i ^ j;
        if (ixj > i) {
          bool up = ((i & k) == 0);
          float s1 = ss[i], s2 = ss[ixj];
          int i1 = si[i], i2 = si[ixj];
          bool firstBefore = (s1 > s2) || (s1 == s2 && i1 < i2);
          if (up ? !firstBefore : firstBefore) {
            ss[i] = s2; ss[ixj] = s1; si[i] = i2; si[ixj] = i1;
          }
        }
      }
      __syncthreads();
    }
  }
  if (tid == 0) {                       // stable partition of keep[0..K-1]
    int na = 0, ni = 0;
    for (int j = 0; j < KKEEP_; ++j) {
      int v = si[j];
      if (v < NAUD_) lista[na++] = v;
      else           listi[ni++] = v - NAUD_;
    }
    s_na = na;
  }
  __syncthreads();
  const int na = s_na;
  const int ni = KKEEP_ - na;
  for (int j = tid; j < NTA_; j += 256) {
    pol_a[(size_t)b * NTA_ + j] = (j == 0) ? 1.f : ((j - 1 < na) ? 1.f : 0.f);
    ga[(size_t)b * NTA_ + j] = (j == 0) ? 0 : ((j - 1 < na) ? lista[j - 1] + 1 : 1);
  }
  for (int j = tid; j < NTI_; j += 256) {
    pol_i[(size_t)b * NTI_ + j] = (j == 0) ? 1.f : ((j - 1 < ni) ? 1.f : 0.f);
    gi[(size_t)b * NTI_ + j] = (j == 0) ? 0 : ((j - 1 < ni) ? listi[j - 1] + 1 : 1);
  }
}

__global__ void k_gather(const float* __restrict__ src, const int* __restrict__ gidx,
                         float* __restrict__ dst, int Nt, int Ssrc)
{
  int j = blockIdx.x, b = blockIdx.y;
  int g = gidx[(size_t)b * Nt + j];
  const float* s = src + ((size_t)b * Ssrc + g) * D_;
  float* d = dst + ((size_t)b * Nt + j) * D_;
  for (int c = threadIdx.x; c < D_ / 4; c += 256)
    ((v4f*)d)[c] = ((const v4f*)s)[c];
}

__global__ void k_assemble_feats(const float* __restrict__ aln,
                                 const float* __restrict__ iln,
                                 float* __restrict__ feats)
{
  int j = blockIdx.x, b = blockIdx.y;
  const float* src = (j < KKEEP_)
      ? aln + ((size_t)b * NTA_ + 1 + j) * D_
      : iln + ((size_t)b * NTI_ + 1 + (j - KKEEP_)) * D_;
  float* dst = feats + ((size_t)b * FEATN_ + j) * D_;
  for (int c = threadIdx.x; c < D_ / 4; c += 256)
    ((v4f*)dst)[c] = ((const v4f*)src)[c];
}

__global__ void k_cls(const float* __restrict__ aln, const float* __restrict__ iln,
                      float* __restrict__ cls)
{
  int b = blockIdx.x;
  for (int c = threadIdx.x; c < 2 * D_; c += 256)
    cls[(size_t)b * 2 * D_ + c] =
        (c < D_) ? aln[(size_t)b * NTA_ * D_ + c]
                 : iln[(size_t)b * NTI_ * D_ + (c - D_)];
}

// ---------------------------------------------------------------------------
// Host side
// ---------------------------------------------------------------------------
static void launch_gemm(const float* A, const float* W, const float* bias,
                        const float* res, float* C, int M, int N, int K,
                        int act, hipStream_t s)
{
  dim3 g((N + 127) / 128, (M + 127) / 128);
  k_gemm<<<g, dim3(256), 0, s>>>(A, W, bias, res, C, M, N, K, act);
}

// w[]: ln1_g, ln1_b, wqkv, bqkv, wproj, bproj, ln2_g, ln2_b, wfc1, bfc1, wfc2, bfc2
static void block_pass(float* x, const float* pol, const float* const* w,
                       int Nt, int Npad, float* buf1, float* buf2, float* buf3,
                       float* buf4, hipStream_t s)
{
  const int M = B_ * Nt;
  k_layernorm<<<dim3(M), dim3(256), 0, s>>>(x, w[0], w[1], buf1, D_);
  launch_gemm(buf1, w[2], w[3], nullptr, buf2, M, 3 * D_, D_, 0, s);
  size_t shm = (size_t)32 * 64 * 2 + (size_t)Npad * 448;     // <= 228KB (LDS 320KB/WGP)
  k_attn<<<dim3((Nt + 31) / 32, H_, B_), dim3(256), shm, s>>>(buf2, pol, buf4, Nt, Npad);
  launch_gemm(buf4, w[4], w[5], x, x, M, D_, D_, 0, s);      // proj + residual
  k_layernorm<<<dim3(M), dim3(256), 0, s>>>(x, w[6], w[7], buf1, D_);
  launch_gemm(buf1, w[8], w[9], nullptr, buf3, M, 4 * D_, D_, 1, s);   // fc1+gelu
  launch_gemm(buf3, w[10], w[11], x, x, M, D_, 4 * D_, 0, s);          // fc2 + residual
}

extern "C" void kernel_launch(void* const* d_in, const int* in_sizes, int n_in,
                              void* d_out, int out_size, void* d_ws, size_t ws_size,
                              hipStream_t stream)
{
  (void)in_sizes; (void)n_in; (void)out_size; (void)ws_size;
  const float* audio = (const float*)d_in[0];
  const float* image = (const float*)d_in[1];
  int ii = 2;
  auto nx = [&]() { return (const float*)d_in[ii++]; };
  // params flattened in dict-insertion order: pred, blk_a, blk_i, norms, head
  const float* p_ln_g = nx(); const float* p_ln_b = nx();
  const float* p_w_in = nx(); const float* p_b_in = nx();
  const float* p_w1 = nx();   const float* p_b1 = nx();
  const float* p_w2 = nx();   const float* p_b2 = nx();
  const float* p_w3 = nx();   const float* p_b3 = nx();
  const float* WA[12]; for (int j = 0; j < 12; ++j) WA[j] = nx();
  const float* WI[12]; for (int j = 0; j < 12; ++j) WI[j] = nx();
  const float* norm_a_g = nx(); const float* norm_a_b = nx();
  const float* norm_i_g = nx(); const float* norm_i_b = nx();
  const float* head_ln_g = nx(); const float* head_ln_b = nx();
  const float* head_w = nx();   const float* head_b = nx();

  float* ws = (float*)d_ws;
  size_t off = 0;
  float* spatial = ws + off; off += (size_t)B_ * NP_ * D_;
  float* buf1    = ws + off; off += (size_t)B_ * NP_ * D_;        // >= final LNs
  float* buf2    = ws + off; off += (size_t)B_ * NTA_ * 3 * D_;   // qkv / pred-h
  float* buf3    = ws + off; off += (size_t)B_ * NTA_ * 4 * D_;   // ffn / h3
  float* buf4    = ws + off; off += (size_t)B_ * NTA_ * D_;       // attn-o / h4
  float* y2      = ws + off; off += (size_t)B_ * NP_ * 2;
  float* score   = ws + off; off += (size_t)B_ * NP_;
  float* gpool   = ws + off; off += (size_t)B_ * (D_ / 2);
  float* pol_a   = ws + off; off += (size_t)B_ * NTA_;
  float* pol_i   = ws + off; off += (size_t)B_ * NTI_;
  int* ga = (int*)(ws + off); off += (size_t)B_ * NTA_;
  int* gi = (int*)(ws + off); off += (size_t)B_ * NTI_;
  float* ax      = ws + off; off += (size_t)B_ * NTA_ * D_;
  float* ix      = ws + off; off += (size_t)B_ * NTI_ * D_;
  float* cls     = ws + off; off += (size_t)B_ * 2 * D_;
  float* clsln   = ws + off; off += (size_t)B_ * 2 * D_;

  const dim3 thr(256);
  // ---- predictor ----
  k_build_spatial<<<dim3(NP_, B_), thr, 0, stream>>>(audio, image, spatial);
  k_layernorm<<<dim3(B_ * NP_), thr, 0, stream>>>(spatial, p_ln_g, p_ln_b, buf1, D_);
  launch_gemm(buf1, p_w_in, p_b_in, nullptr, buf2, B_ * NP_, D_, D_, 1, stream);
  k_pool<<<dim3(D_ / 2, B_), thr, 0, stream>>>(buf2, gpool);
  k_build_h2<<<dim3(NP_, B_), thr, 0, stream>>>(buf2, gpool, buf1);
  launch_gemm(buf1, p_w1, p_b1, nullptr, buf3, B_ * NP_, D_ / 2, D_, 1, stream);
  launch_gemm(buf3, p_w2, p_b2, nullptr, buf4, B_ * NP_, D_ / 4, D_ / 2, 1, stream);
  launch_gemm(buf4, p_w3, p_b3, nullptr, y2, B_ * NP_, 2, D_ / 4, 0, stream);
  k_score<<<dim3((B_ * NP_ + 255) / 256), thr, 0, stream>>>(y2, score);
  k_topk<<<dim3(B_), thr, 0, stream>>>(score, pol_a, pol_i, ga, gi);
  k_gather<<<dim3(NTA_, B_), thr, 0, stream>>>(audio, ga, ax, NTA_, NAUD_ + 1);
  k_gather<<<dim3(NTI_, B_), thr, 0, stream>>>(image, gi, ix, NTI_, NIMG_ + 1);

  // ---- 3 transformer blocks per modality ----
  for (int it = 0; it < 3; ++it) {
    block_pass(ax, pol_a, WA, NTA_, 512, buf1, buf2, buf3, buf4, stream);
    block_pass(ix, pol_i, WI, NTI_, 256, buf1, buf2, buf3, buf4, stream);
  }

  // ---- final norms, feats, head ----
  float* aln = buf1;
  float* iln = buf1 + (size_t)B_ * NTA_ * D_;
  k_layernorm<<<dim3(B_ * NTA_), thr, 0, stream>>>(ax, norm_a_g, norm_a_b, aln, D_);
  k_layernorm<<<dim3(B_ * NTI_), thr, 0, stream>>>(ix, norm_i_g, norm_i_b, iln, D_);

  float* out = (float*)d_out;
  float* feats = out + (size_t)B_ * NCLS_;
  k_assemble_feats<<<dim3(FEATN_, B_), thr, 0, stream>>>(aln, iln, feats);
  k_cls<<<dim3(B_), thr, 0, stream>>>(aln, iln, cls);
  k_layernorm<<<dim3(B_), thr, 0, stream>>>(cls, head_ln_g, head_ln_b, clsln, 2 * D_);
  launch_gemm(clsln, head_w, head_b, nullptr, out, B_, NCLS_, 2 * D_, 0, stream);
}
